// PNAConvModel_15625091023067
// MI455X (gfx1250) — matrix operand order repaired and checked by reading, not compiled, not run on hardware
//
#include <hip/hip_runtime.h>
#include <hip/hip_bf16.h>
#include <math.h>

// ---------------- types for gfx1250 WMMA ----------------
typedef __bf16 bf16_t;
typedef __attribute__((ext_vector_type(16))) __bf16    v16bf;
typedef __attribute__((ext_vector_type(8)))  float     v8f;
typedef __attribute__((ext_vector_type(4)))  unsigned  u32x4;

#define NN    50000
#define EE    800000
#define FIN   64
#define HH    128
#define OUTD  64
#define AVG_LOG 1.6378164280770893f
#define EPS_BN  1e-5f
#define EPS_STD 1e-5f

// round-to-nearest-even f32 -> bf16
__device__ __forceinline__ bf16_t f2bf(float f) {
  unsigned b = __float_as_uint(f);
  b += 0x7FFFu + ((b >> 16) & 1u);
  unsigned short r = (unsigned short)(b >> 16);
  union { unsigned short u; bf16_t b; } c; c.u = r; return c.b;
}
// order-preserving float <-> uint for atomicMin/Max
__device__ __forceinline__ unsigned fenc(float f) {
  unsigned b = __float_as_uint(f);
  return (b & 0x80000000u) ? ~b : (b | 0x80000000u);
}
__device__ __forceinline__ float fdec(unsigned e) {
  unsigned b = (e & 0x80000000u) ? (e ^ 0x80000000u) : ~e;
  return __uint_as_float(b);
}

// ---------------- utility kernels ----------------
__global__ void k_fillf(float* p, size_t n, float v) {
  size_t i = (size_t)blockIdx.x * blockDim.x + threadIdx.x;
  if (i < n) p[i] = v;
}
__global__ void k_fillu(unsigned* p, size_t n, unsigned v) {
  size_t i = (size_t)blockIdx.x * blockDim.x + threadIdx.x;
  if (i < n) p[i] = v;
}
__global__ void k_cvt_f2bf(const float* __restrict__ s, bf16_t* __restrict__ d, size_t n) {
  size_t i = (size_t)blockIdx.x * blockDim.x + threadIdx.x;
  if (i < n) d[i] = f2bf(s[i]);
}

// ---------------- edge scatter (bandwidth/atomic bound phase) ----------------
__global__ void k_scatter(const float* __restrict__ x, const int* __restrict__ src,
                          const int* __restrict__ dst, float* __restrict__ deg,
                          float* __restrict__ sum, float* __restrict__ sum2,
                          unsigned* __restrict__ mnE, unsigned* __restrict__ mxE,
                          int F) {
  size_t i = (size_t)blockIdx.x * blockDim.x + threadIdx.x;
  size_t total = (size_t)EE * F;
  if (i >= total) return;
  int e = (int)(i / F);
  int f = (int)(i % F);
  int s = src[e], d = dst[e];
  float m = x[(size_t)s * F + f];
  size_t o = (size_t)d * F + f;
  atomicAdd(&sum[o],  m);
  atomicAdd(&sum2[o], m * m);
  atomicMin(&mnE[o], fenc(m));
  atomicMax(&mxE[o], fenc(m));
  if (f == 0) atomicAdd(&deg[d], 1.0f);
}

// ---------------- PNA finalize -> [N, 12F] bf16 ----------------
__global__ void k_finalize(const float* __restrict__ deg, const float* __restrict__ sum,
                           const float* __restrict__ sum2, const unsigned* __restrict__ mnE,
                           const unsigned* __restrict__ mxE, bf16_t* __restrict__ agg, int F) {
  size_t i = (size_t)blockIdx.x * blockDim.x + threadIdx.x;
  size_t total = (size_t)NN * F;
  if (i >= total) return;
  int n = (int)(i / F);
  int f = (int)(i % F);
  float d    = deg[n];
  float degc = fmaxf(d, 1.0f);
  float mean = sum[i] / degc;
  float var  = fmaxf(sum2[i] / degc - mean * mean, 0.0f);
  float stdv = sqrtf(var + EPS_STD);
  float mn = (d > 0.0f) ? fdec(mnE[i]) : 0.0f;
  float mx = (d > 0.0f) ? fdec(mxE[i]) : 0.0f;
  float logd = logf(degc + 1.0f);
  float amp = logd / AVG_LOG;
  float att = AVG_LOG / logd;
  bf16_t* row = agg + (size_t)n * 12 * F;
  row[0*F + f]  = f2bf(mean);       row[1*F + f]  = f2bf(mn);
  row[2*F + f]  = f2bf(mx);         row[3*F + f]  = f2bf(stdv);
  row[4*F + f]  = f2bf(mean * amp); row[5*F + f]  = f2bf(mn * amp);
  row[6*F + f]  = f2bf(mx * amp);   row[7*F + f]  = f2bf(stdv * amp);
  row[8*F + f]  = f2bf(mean * att); row[9*F + f]  = f2bf(mn * att);
  row[10*F + f] = f2bf(mx * att);   row[11*F + f] = f2bf(stdv * att);
}

// ---------------- generic bf16 WMMA GEMM with async B-panel staging ----------
// A [M x K] row-major bf16, B [K x Nc] row-major bf16, C = A*B + bias.
// One wave per 16(M) x 64(N) tile; 8 waves / block along M. All 8 waves share
// one 32x64 bf16 B k-panel staged in LDS via GLOBAL_LOAD_ASYNC_TO_LDS_B128
// (one B128 per thread, 256 threads == 4KB panel), synced with
// s_wait_asynccnt + workgroup barrier.
// A fragment (ISA 7.12.2, 16-bit A 16x32): lane&15 = M row, lane>>4 selects the
// K-half; elements 0..7 -> K = kb + 8*(lane>>4) + e, elements 8..15 -> +16.
// B fragment: lane&15 = N column, elements in same K order as A (from LDS).
// C/D: column = nbase + (lane&15), rows = mbase + r + 8*(lane>>4).
__global__ void __launch_bounds__(256)
k_gemm_bf16(const bf16_t* __restrict__ A, const bf16_t* __restrict__ B,
            const float* __restrict__ bias, float* __restrict__ outf,
            bf16_t* __restrict__ outbf, int M, int K, int Nc, int relu) {
  __shared__ bf16_t btile[32 * 64];
  const int tid    = threadIdx.x;
  const int lane   = tid & 31;
  const int wave   = tid >> 5;
  const int mtiles = M / 16;
  int mtile = blockIdx.x * 8 + wave;
  const bool active = (mtile < mtiles);
  if (!active) mtile = mtiles - 1;   // clamp: keep wave alive for barriers/WMMA
  const int mbase  = mtile * 16;
  const int nblock = blockIdx.y * 64;

  // B-copy mapping: thread -> (row, 8-element column chunk) of the 32x64 panel
  const int brow = tid >> 3;
  const int bcol = (tid & 7) * 8;
  const uint32_t ldsAddr = (uint32_t)(uintptr_t)(btile + brow * 64 + bcol);
  const bf16_t* bsrc = B + (size_t)brow * Nc + nblock + bcol;

  const bf16_t* arow = A + (size_t)(mbase + (lane & 15)) * K + (lane >> 4) * 8;
  const bf16_t* bfrag = btile + (lane >> 4) * 8 * 64 + (lane & 15);

  v8f acc[4] = {};
  for (int kb = 0; kb < K; kb += 32) {
    // stage B panel: global -> LDS (async, CDNA5 path)
    uint64_t gaddr = (uint64_t)(uintptr_t)(bsrc + (size_t)kb * Nc);
    asm volatile("global_load_async_to_lds_b128 %0, %1, off"
                 :: "v"(ldsAddr), "v"(gaddr) : "memory");
    asm volatile("s_wait_asynccnt 0x0" ::: "memory");
    __syncthreads();

    if (kb + 64 < K) __builtin_prefetch(arow + kb + 64, 0, 3);
    union { u32x4 u[2]; v16bf v; } af;
    af.u[0] = *(const u32x4*)(arow + kb);
    af.u[1] = *(const u32x4*)(arow + kb + 16);
#pragma unroll
    for (int nt = 0; nt < 4; ++nt) {
      union { bf16_t e[16]; v16bf v; } bfm;
#pragma unroll
      for (int s = 0; s < 16; ++s)
        bfm.e[s] = bfrag[((s & 7) + ((s >> 3) << 4)) * 64 + nt * 16];
      acc[nt] = __builtin_amdgcn_wmma_f32_16x16x32_bf16(
          false, af.v, false, bfm.v, (short)0, acc[nt], false, false);
    }
    __syncthreads();   // panel fully consumed before next overwrite
  }

  if (!active) return;
  const int mr0 = mbase + ((lane >> 4) << 3);
#pragma unroll
  for (int nt = 0; nt < 4; ++nt) {
    int n = nblock + nt * 16 + (lane & 15);
    float bv = bias ? bias[n] : 0.0f;
#pragma unroll
    for (int r = 0; r < 8; ++r) {
      float v = acc[nt][r] + bv;
      if (relu) v = fmaxf(v, 0.0f);
      size_t idx = (size_t)(mr0 + r) * Nc + n;
      if (outf)  outf[idx]  = v;
      if (outbf) outbf[idx] = f2bf(v);
    }
  }
}

// ---------------- BN column stats (coalesced + LDS + atomics) ----------------
__global__ void k_colreduce(const float* __restrict__ y, float* __restrict__ gsums) {
  __shared__ float ls[256], ls2[256];
  int c = threadIdx.x & (HH - 1);
  int rowsPerBlock = (NN + gridDim.x - 1) / gridDim.x;
  int r0 = blockIdx.x * rowsPerBlock;
  int r1 = min(r0 + rowsPerBlock, NN);
  float s = 0.0f, s2 = 0.0f;
  for (int r = r0 + (threadIdx.x >> 7); r < r1; r += 2) {
    float v = y[(size_t)r * HH + c];
    s += v; s2 += v * v;
  }
  ls[threadIdx.x] = s; ls2[threadIdx.x] = s2;
  __syncthreads();
  if (threadIdx.x < HH) {
    atomicAdd(&gsums[c],      ls[threadIdx.x] + ls[threadIdx.x + HH]);
    atomicAdd(&gsums[HH + c], ls2[threadIdx.x] + ls2[threadIdx.x + HH]);
  }
}

__global__ void k_bn_apply(const float* __restrict__ y, const float* __restrict__ gsums,
                           const float* __restrict__ gamma, const float* __restrict__ beta,
                           bf16_t* __restrict__ ybf) {
  size_t i = (size_t)blockIdx.x * blockDim.x + threadIdx.x;
  if (i >= (size_t)NN * HH) return;
  int c = (int)(i % HH);
  float mean = gsums[c] * (1.0f / NN);
  float var  = gsums[HH + c] * (1.0f / NN) - mean * mean;
  float v = gamma[c] * (y[i] - mean) * rsqrtf(var + EPS_BN) + beta[c];
  ybf[i] = f2bf(v);
}

// ---------------- GRU gates (torch order r,z,n) ----------------
__global__ void k_gru(const float* __restrict__ gi, const float* __restrict__ gh,
                      float* __restrict__ h, bf16_t* __restrict__ hbf) {
  size_t i = (size_t)blockIdx.x * blockDim.x + threadIdx.x;
  if (i >= (size_t)NN * HH) return;
  int n = (int)(i / HH);
  int j = (int)(i % HH);
  const float* gin = gi + (size_t)n * 3 * HH;
  const float* ghn = gh + (size_t)n * 3 * HH;
  float r  = 1.0f / (1.0f + expf(-(gin[j] + ghn[j])));
  float z  = 1.0f / (1.0f + expf(-(gin[HH + j] + ghn[HH + j])));
  float nn = tanhf(gin[2 * HH + j] + r * ghn[2 * HH + j]);
  float hv = (1.0f - z) * nn + z * h[i];
  h[i]   = hv;
  hbf[i] = f2bf(hv);
}

// ---------------- host orchestration ----------------
static inline size_t rnd256(size_t b) { return (b + 255) & ~(size_t)255; }

extern "C" void kernel_launch(void* const* d_in, const int* in_sizes, int n_in,
                              void* d_out, int out_size, void* d_ws, size_t ws_size,
                              hipStream_t stream) {
  (void)in_sizes; (void)n_in; (void)out_size; (void)ws_size;
  const float* x_in   = (const float*)d_in[0];
  const int*   ei     = (const int*)d_in[1];
  const int*   srcI   = ei;
  const int*   dstI   = ei + EE;
  const float* conv_w[3] = { (const float*)d_in[2], (const float*)d_in[4], (const float*)d_in[6] };
  const float* conv_b[3] = { (const float*)d_in[3], (const float*)d_in[5], (const float*)d_in[7] };
  const float* bn_gamma = (const float*)d_in[8];
  const float* bn_beta  = (const float*)d_in[9];
  const float* fc_w     = (const float*)d_in[10];
  const float* fc_b     = (const float*)d_in[11];
  const float* w_ih     = (const float*)d_in[12];
  const float* w_hh     = (const float*)d_in[13];
  const float* b_ih     = (const float*)d_in[14];
  const float* b_hh     = (const float*)d_in[15];
  const float* last_w   = (const float*)d_in[16];
  const float* last_b   = (const float*)d_in[17];

  // workspace arena (~346 MB)
  char* w = (char*)d_ws;
  auto alloc = [&](size_t bytes) -> char* { char* p = w; w += rnd256(bytes); return p; };
  float*    deg   = (float*)   alloc((size_t)NN * 4);
  float*    sum   = (float*)   alloc((size_t)NN * HH * 4);
  float*    sum2  = (float*)   alloc((size_t)NN * HH * 4);
  unsigned* mnE   = (unsigned*)alloc((size_t)NN * HH * 4);
  unsigned* mxE   = (unsigned*)alloc((size_t)NN * HH * 4);
  bf16_t*   agg12 = (bf16_t*)  alloc((size_t)NN * 12 * HH * 2);  // aliased by gi/gh later
  float*    gi    = (float*)agg12;                               // N*384 f32 = 76.8MB
  float*    gh    = gi + (size_t)NN * 3 * HH;                    // second half of agg12
  float*    y     = (float*)   alloc((size_t)NN * HH * 4);
  bf16_t*   ybf   = (bf16_t*)  alloc((size_t)NN * HH * 2);
  bf16_t*   x2bf  = (bf16_t*)  alloc((size_t)NN * HH * 2);
  float*    hbuf  = (float*)   alloc((size_t)NN * HH * 4);
  bf16_t*   hbf   = (bf16_t*)  alloc((size_t)NN * HH * 2);
  float*    gsums = (float*)   alloc(2 * HH * 4);
  bf16_t*   convw_bf = (bf16_t*)alloc((size_t)12 * HH * HH * 2);
  bf16_t*   fcw_bf   = (bf16_t*)alloc((size_t)HH * HH * 2);
  bf16_t*   wih_bf   = (bf16_t*)alloc((size_t)HH * 3 * HH * 2);
  bf16_t*   whh_bf   = (bf16_t*)alloc((size_t)HH * 3 * HH * 2);
  bf16_t*   lastw_bf = (bf16_t*)alloc((size_t)HH * OUTD * 2);

  auto blk = [](size_t n) { return dim3((unsigned)((n + 255) / 256)); };
  auto fillf = [&](float* p, size_t n, float v) { k_fillf<<<blk(n), 256, 0, stream>>>(p, n, v); };
  auto fillu = [&](unsigned* p, size_t n, unsigned v) { k_fillu<<<blk(n), 256, 0, stream>>>(p, n, v); };
  auto cvt = [&](const float* s, bf16_t* d, size_t n) { k_cvt_f2bf<<<blk(n), 256, 0, stream>>>(s, d, n); };
  auto gemm = [&](const bf16_t* A, const bf16_t* B, const float* bias, float* outf,
                  bf16_t* outbf, int M, int K, int Nc, int relu) {
    dim3 g((unsigned)((M / 16 + 7) / 8), (unsigned)(Nc / 64));
    k_gemm_bf16<<<g, 256, 0, stream>>>(A, B, bias, outf, outbf, M, K, Nc, relu);
  };

  // h = 0, hbf = 0; convert shared weights once
  fillf(hbuf, (size_t)NN * HH, 0.0f);
  fillf((float*)hbf, (size_t)NN * HH / 2, 0.0f);
  cvt(w_ih, wih_bf, (size_t)HH * 3 * HH);
  cvt(w_hh, whh_bf, (size_t)HH * 3 * HH);
  cvt(last_w, lastw_bf, (size_t)HH * OUTD);

  for (int l = 0; l < 3; ++l) {
    const int F  = (l == 0) ? FIN : HH;
    const int Kc = 12 * F;
    const float* xcur = (l == 0) ? x_in : hbuf;

    // 1) init aggregation buffers
    fillf(deg, NN, 0.0f);
    fillf(sum,  (size_t)NN * F, 0.0f);
    fillf(sum2, (size_t)NN * F, 0.0f);
    fillu(mnE,  (size_t)NN * F, 0xFFFFFFFFu);
    fillu(mxE,  (size_t)NN * F, 0u);

    // 2) edge scatter
    k_scatter<<<blk((size_t)EE * F), 256, 0, stream>>>(xcur, srcI, dstI, deg, sum, sum2, mnE, mxE, F);

    // 3) finalize -> agg12 [N, 12F] bf16
    k_finalize<<<blk((size_t)NN * F), 256, 0, stream>>>(deg, sum, sum2, mnE, mxE, agg12, F);

    // 4) conv post-linear (WMMA): y = agg12 @ conv_w + conv_b
    cvt(conv_w[l], convw_bf, (size_t)Kc * HH);
    gemm(agg12, convw_bf, conv_b[l], y, nullptr, NN, Kc, HH, 0);

    // 5) batchnorm (training-mode batch stats) -> ybf
    fillf(gsums, 2 * HH, 0.0f);
    k_colreduce<<<128, 256, 0, stream>>>(y, gsums);
    k_bn_apply<<<blk((size_t)NN * HH), 256, 0, stream>>>(y, gsums, bn_gamma + l * HH,
                                                         bn_beta + l * HH, ybf);

    // 6) FC + ReLU (WMMA) -> x2bf
    cvt(fc_w + (size_t)l * HH * HH, fcw_bf, (size_t)HH * HH);
    gemm(ybf, fcw_bf, fc_b + l * HH, nullptr, x2bf, NN, HH, HH, 1);

    // 7) GRU: gi = x2 @ w_ih + b_ih ; gh = h @ w_hh + b_hh ; gates
    gemm(x2bf, wih_bf, b_ih, gi, nullptr, NN, HH, 3 * HH, 0);
    gemm(hbf,  whh_bf, b_hh, gh, nullptr, NN, HH, 3 * HH, 0);
    k_gru<<<blk((size_t)NN * HH), 256, 0, stream>>>(gi, gh, hbuf, hbf);
  }

  // final projection: out = h @ last_w + last_b
  gemm(hbf, lastw_bf, last_b, (float*)d_out, nullptr, NN, HH, OUTD, 0);
}
